// SINDy_layer_29334626632132
// MI455X (gfx1250) — matrix operand order repaired and checked by reading, not compiled, not run on hardware
//
#include <hip/hip_runtime.h>

#define ZDIM    32
#define NTERMS  6545
#define ZSTRIDE 17        // zT padded row stride (floats)
#define NPAIRS  561       // pairs (i<=j) over augmented 33-dim z~ (q = j(j+1)/2 + i)
#define QPAD    576       // padded pair count (18 chunks of 32 max, at k=32)
#define PSTRIDE 584       // PPT per-row stride in halfs (16B-aligned, bank-friendly)
#define NCBLK   221       // sum over k=0..32 of ceil((k+1)(k+2)/2 / 32)
#define ONE_IDX 32        // z~[32] = 1.0f

typedef __attribute__((ext_vector_type(16))) _Float16 v16h;
typedef __attribute__((ext_vector_type(8)))  float    v8f;

// ---------------------------------------------------------------------------
// Setup kernel: W in f16 A-fragment layout, one (chunk, m-tile) block = 512 halfs.
// Chunk blocks enumerated (k asc, c asc): for outer k, K-dim = prefix of pairs
// q < Q(k) = (k+1)(k+2)/2 (j-major order makes the j<=k mask a contiguous prefix).
// W_k[q(i,j), m] = Xi_eff[t(i,j,k), m]; padded slots (q >= Q(k)) are 0.
// Reference term order: [bias][linear 32][quad 528 i<=j][cubic 5984 i<=j<=k].
// A-fragment layout per ISA 7.12.2 (16-bit A 16x32): lane L<16 holds K={0..7,16..23},
// L>=16 holds K={8..15,24..31}; VGPR v packs (K=2v,2v+1) lo/hi.
// ---------------------------------------------------------------------------
__global__ void sindy_build_wA_kernel(const float* __restrict__ Xi,
                                      const float* __restrict__ Xi_mask,
                                      _Float16* __restrict__ wA) {
    int idx = blockIdx.x * blockDim.x + threadIdx.x;   // over NCBLK*32*32
    if (idx >= NCBLK * 1024) return;
    int cblk = idx >> 10;        // global chunk block
    int kk   = (idx >> 5) & 31;  // K position within chunk
    int m    = idx & 31;         // output z-dim

    // locate (kOuter, c) for this chunk block
    int kOuter = 0, base = 0;
    for (;;) {
        int Q   = (kOuter + 1) * (kOuter + 2) / 2;
        int nch = (Q + 31) >> 5;
        if (cblk < base + nch) break;
        base += nch;
        ++kOuter;
    }
    int c  = cblk - base;
    int q  = c * 32 + kk;
    int Qk = (kOuter + 1) * (kOuter + 2) / 2;

    float v = 0.0f;
    if (q < Qk) {
        // decode q -> (i,j), j-major: q = j(j+1)/2 + i, i<=j
        int j = 0;
        while ((j + 1) * (j + 2) / 2 <= q) ++j;
        int i = q - j * (j + 1) / 2;
        // reference term index t(i, j, kOuter) over augmented z~
        int t;
        if (kOuter == ONE_IDX) {
            if (j == ONE_IDX) t = (i == ONE_IDX) ? 0 : 1 + i;             // bias / linear
            else              t = 33 + (i * ZDIM - (i * (i - 1)) / 2 + (j - i)); // quad
        } else {
            int off = 0;                                                  // cubic
            for (int a = 0; a < i; ++a) off += (ZDIM - a) * (ZDIM - a + 1) / 2;
            for (int b2 = i; b2 < j; ++b2) off += (ZDIM - b2);
            t = 561 + off + (kOuter - j);
        }
        v = Xi[t * ZDIM + m] * Xi_mask[t * ZDIM + m];
    }

    int mt = m >> 4, mrow = m & 15;
    int baseK = (kk >= 16) ? 16 : 0;
    int kk2 = kk - baseK, halfSel = kk2 >> 3, kk3 = kk2 & 7;
    int vslot = (baseK >> 2) + (kk3 >> 1), pairIdx = kk3 & 1;
    int laneL = mrow + 16 * halfSel;

    wA[((size_t)(cblk * 2 + mt)) * 512 + (size_t)laneL * 16 + vslot * 2 + pairIdx] =
        (_Float16)v;
}

// ---------------------------------------------------------------------------
// Main kernel: 1024 blocks x 128 threads (4 waves), wave handles 16 batch rows.
// For each outer k: S_k(32x16) = W_k^T (A, two 16x32 f16 tiles) x PP-prefix^T
// (B, straight contiguous LDS loads); then dz += z~_k * S_k (f32).
// ---------------------------------------------------------------------------
__global__ __launch_bounds__(128) void sindy_wmma_kernel(
    const float*    __restrict__ z,
    const _Float16* __restrict__ wA,
    float*          __restrict__ out) {
    __shared__ float zT[4][34 * ZSTRIDE];                     // +1 pad index row
    __shared__ __align__(32) _Float16 ppt[4][16 * PSTRIDE];   // PPT[row][q], transposed

    const int tid   = threadIdx.x;
    const int wave  = tid >> 5;
    const int lane  = tid & 31;
    const int nrow  = lane & 15;            // batch row within tile == D column n
    const int halfU = lane >> 4;
    const long rBase = (long)blockIdx.x * 64 + (long)wave * 16;

    float*    zw = &zT[wave][0];
    _Float16* pw = &ppt[wave][0];

    // Stage z~ transposed into LDS: zT[k*17 + row]
    {
        const int kbase = halfU * 16;
        const float4* zp = (const float4*)(z + (rBase + nrow) * ZDIM + kbase);
        float4 q0 = zp[0], q1 = zp[1], q2 = zp[2], q3 = zp[3];
        float vals[16] = {q0.x, q0.y, q0.z, q0.w, q1.x, q1.y, q1.z, q1.w,
                          q2.x, q2.y, q2.z, q2.w, q3.x, q3.y, q3.z, q3.w};
#pragma unroll
        for (int i = 0; i < 16; ++i)
            zw[(kbase + i) * ZSTRIDE + nrow] = vals[i];
        if (halfU == 0) zw[ONE_IDX * ZSTRIDE + nrow] = 1.0f;
    }
    __syncthreads();

    // Build PPT[row][q] = z~_i * z~_j, q-major (q = j(j+1)/2 + i), indices kept
    // incrementally in registers (no tables, no global loads). Half-waves cover
    // alternate q; uniform 281 trips (upper half's phantom q=561 lands in pad,
    // overwritten by the zero-pad loop below).
    {
        int i = 0, jj = halfU;
        int qAddr = nrow * PSTRIDE + halfU;
        for (int it = 0; it < 281; ++it) {
            float vi = zw[i * ZSTRIDE + nrow];
            float vj = zw[jj * ZSTRIDE + nrow];
            pw[qAddr] = (_Float16)(vi * vj);
            qAddr += 2;
            i += 2;
            int e = i - (jj + 1);
            if (e >= 0) { jj += 1; i = e; }   // single wrap always sufficient
        }
        // zero pad q in [561, 576) for all 16 rows (NaN-safe padded WMMA columns)
        for (int x = lane; x < 16 * (QPAD - NPAIRS); x += 32) {
            int row = x / (QPAD - NPAIRS);
            int qq  = NPAIRS + x % (QPAD - NPAIRS);
            pw[row * PSTRIDE + qq] = (_Float16)0.0f;
        }
    }
    __syncthreads();

    v8f dz0 = {};   // dz tile m=0..15
    v8f dz1 = {};   // dz tile m=16..31
    const int colSel = halfU << 3;   // B layout: lanes 0-15 hold K {0..7,16..23}, lanes 16-31 K {8..15,24..31}
    const _Float16* pb = pw + nrow * PSTRIDE + colSel;
    const _Float16* ap = wA + (size_t)lane * 16;

    // prefetch first A chunk (wA is padded by one 2KB block for the final prefetch)
    v16h a0 = *(const v16h*)(ap);
    v16h a1 = *(const v16h*)(ap + 512);
    ap += 1024;

    for (int k = 0; k <= 32; ++k) {
        const int nch = (((k + 1) * (k + 2) / 2) + 31) >> 5;
        v8f acc0 = {};
        v8f acc1 = {};
        for (int c = 0; c < nch; ++c) {
            v16h na0 = *(const v16h*)(ap);        // prefetch next chunk's A tiles
            v16h na1 = *(const v16h*)(ap + 512);
            ap += 1024;
            typedef __attribute__((ext_vector_type(8))) _Float16 v8h_t;
            v8h_t blo = *(const v8h_t*)(pb + c * 32);
            v8h_t bhi = *(const v8h_t*)(pb + c * 32 + 16);
            v16h b = __builtin_shufflevector(blo, bhi, 0, 1, 2, 3, 4, 5, 6, 7, 8, 9, 10, 11, 12, 13, 14, 15);

            acc0 = __builtin_amdgcn_wmma_f32_16x16x32_f16(false, a0, false, b,
                                                          (short)0, acc0, false, false);
            acc1 = __builtin_amdgcn_wmma_f32_16x16x32_f16(false, a1, false, b,
                                                          (short)0, acc1, false, false);
            a0 = na0;
            a1 = na1;
        }
        float zk = zw[k * ZSTRIDE + nrow];        // z~_k in f32
#pragma unroll
        for (int r = 0; r < 8; ++r) {
            dz0[r] += zk * acc0[r];
            dz1[r] += zk * acc1[r];
        }
    }

    // D layout: lane column n = nrow (batch row); VGPR r holds M=r (lanes<16) / M=8+r.
    float* op = out + (rBase + nrow) * ZDIM;
    const int mbase = halfU * 8;
    float4 s0; s0.x = dz0[0]; s0.y = dz0[1]; s0.z = dz0[2]; s0.w = dz0[3];
    float4 s1; s1.x = dz0[4]; s1.y = dz0[5]; s1.z = dz0[6]; s1.w = dz0[7];
    float4 s2; s2.x = dz1[0]; s2.y = dz1[1]; s2.z = dz1[2]; s2.w = dz1[3];
    float4 s3; s3.x = dz1[4]; s3.y = dz1[5]; s3.z = dz1[6]; s3.w = dz1[7];
    *(float4*)(op + mbase)          = s0;
    *(float4*)(op + mbase + 4)      = s1;
    *(float4*)(op + 16 + mbase)     = s2;
    *(float4*)(op + 16 + mbase + 4) = s3;
}

// ---------------------------------------------------------------------------
// Workspace: wA = NCBLK*2*512 halfs = 452,608 B, +2 KB prefetch pad.
// ---------------------------------------------------------------------------
extern "C" void kernel_launch(void* const* d_in, const int* in_sizes, int n_in,
                              void* d_out, int out_size, void* d_ws, size_t ws_size,
                              hipStream_t stream) {
    (void)in_sizes; (void)n_in; (void)out_size; (void)ws_size;
    const float* z  = (const float*)d_in[0];
    const float* Xi = (const float*)d_in[1];
    const float* Xm = (const float*)d_in[2];
    float* out = (float*)d_out;

    _Float16* wA = (_Float16*)d_ws;

    sindy_build_wA_kernel<<<(NCBLK * 1024 + 255) / 256, 256, 0, stream>>>(Xi, Xm, wA);
    sindy_wmma_kernel<<<65536 / 64, 128, 0, stream>>>(z, wA, out);
}
